// QuantGPT_67551245631862
// MI455X (gfx1250) — matrix operand order repaired and checked by reading, not compiled, not run on hardware
//
#include <hip/hip_runtime.h>
#include <hip/hip_bf16.h>
#include <stdint.h>

// ---------------------------------------------------------------------------
// QuantGPT block for MI455X (gfx1250): all matmuls via v_wmma_f32_16x16x32_bf16,
// GEMM LDS staging via global_load_async_to_lds_b128 (inline asm, ASYNCcnt
// double buffer) -- the ROCm builtin has a descriptor-style signature, so we
// bypass it per the CDNA5 bridge doc.
// ---------------------------------------------------------------------------

typedef __bf16 bf16_t;
typedef float  v8f   __attribute__((ext_vector_type(8)));
typedef __bf16 v16bf __attribute__((ext_vector_type(16)));
typedef __bf16 v8bf  __attribute__((ext_vector_type(8)));

static constexpr int kT = 2048;   // sequence length
static constexpr int kC = 2048;   // model dim
static constexpr int kH = 16;     // query heads
static constexpr int kG = 4;      // kv heads
static constexpr int kD = 128;    // head dim
static constexpr int kI = 8192;   // mlp dim
static constexpr int kQKV = (kH + 2 * kG) * kD; // 3072

// ---- CDNA5 async memory->LDS path (inline asm; sync fallback off-device) ---
#if defined(__gfx1250__)
#define HAVE_ASYNC_LDS 1
#else
#define HAVE_ASYNC_LDS 0
#endif

#if HAVE_ASYNC_LDS
__device__ __forceinline__ void async_copy16(const void* g, void* l) {
  // 16 bytes per lane, memory -> LDS, tracked by ASYNCcnt.
  // vdst = 32-bit LDS byte address, vaddr = 64-bit global address (GV mode).
  __attribute__((address_space(3))) void* l3 =
      (__attribute__((address_space(3))) void*)l;
  asm volatile("global_load_async_to_lds_b128 %0, %1, off"
               :
               : "v"(l3), "v"(g)
               : "memory");
}
__device__ __forceinline__ void wait_async_le4() {
  asm volatile("s_wait_asynccnt 4" ::: "memory");
}
__device__ __forceinline__ void wait_async_le0() {
  asm volatile("s_wait_asynccnt 0" ::: "memory");
}
#else
__device__ __forceinline__ void wait_async_le4() {}
__device__ __forceinline__ void wait_async_le0() {}
#endif

__device__ __forceinline__ bf16_t f2bf(float f) {
  union { float f; uint32_t u; } a; a.f = f;
  uint32_t u = a.u;
  u += 0x7FFFu + ((u >> 16) & 1u);          // round-to-nearest-even
  union { uint16_t s; bf16_t b; } o;
  o.s = (uint16_t)(u >> 16);
  return o.b;
}

// --------------------------- elementwise kernels ---------------------------

// out = bf16( alpha[c] if (x - beta[c]) > 0 else 0 ), c = i mod `mod` (pow2)
__global__ void hv_to_bf16(const float* __restrict__ x,
                           const float* __restrict__ alpha,
                           const float* __restrict__ beta,
                           bf16_t* __restrict__ out, int n, int mod) {
  int i = blockIdx.x * blockDim.x + threadIdx.x;
  if (i >= n) return;
  int c = i & (mod - 1);
  out[i] = f2bf((x[i] - beta[c] > 0.f) ? alpha[c] : 0.f);
}

__global__ void cvt_to_bf16(const float* __restrict__ x,
                            bf16_t* __restrict__ out, int n) {
  int i = blockIdx.x * blockDim.x + threadIdx.x;
  if (i >= n) return;
  out[i] = f2bf(x[i]);
}

// normed_linear weight prep: per column j (input dim), subtract column mean
// over rows (output dim), divide by column L2 norm, scale; emit bf16 row-major.
__global__ __launch_bounds__(256) void norm_weight(
    const float* __restrict__ w, bf16_t* __restrict__ out,
    int rows, int cols, const float* __restrict__ scale_p) {
  __shared__ float red[256];
  const int j = blockIdx.x;
  const int tid = threadIdx.x;
  float s = 0.f;
  for (int r = tid; r < rows; r += 256) s += w[(size_t)r * cols + j];
  red[tid] = s; __syncthreads();
  for (int o = 128; o > 0; o >>= 1) { if (tid < o) red[tid] += red[tid + o]; __syncthreads(); }
  const float mean = red[0] / (float)rows;
  __syncthreads();
  float q = 0.f;
  for (int r = tid; r < rows; r += 256) {
    float d = w[(size_t)r * cols + j] - mean; q += d * d;
  }
  red[tid] = q; __syncthreads();
  for (int o = 128; o > 0; o >>= 1) { if (tid < o) red[tid] += red[tid + o]; __syncthreads(); }
  const float inv = scale_p[0] / fmaxf(sqrtf(red[0]), 1e-12f);
  for (int r = tid; r < rows; r += 256)
    out[(size_t)r * cols + j] = f2bf((w[(size_t)r * cols + j] - mean) * inv);
}

// RoPE + hv gates on q,k; split qkv(T,3072) into Q(H,T,D), K(G,T,D), Vt(G,D,T)
__global__ __launch_bounds__(128) void rope_split(
    const float* __restrict__ qkv, const float* __restrict__ cosb,
    const float* __restrict__ sinb,
    const float* __restrict__ qa, const float* __restrict__ qb,
    const float* __restrict__ ka, const float* __restrict__ kb,
    bf16_t* __restrict__ Q, bf16_t* __restrict__ Kq, bf16_t* __restrict__ Vt) {
  const int t = blockIdx.x;
  const int j = blockIdx.y;        // 0..23 = (g, slot)
  const int g = j / 6;
  const int slot = j % 6;          // 0..3 q heads, 4 = k, 5 = v
  const int d = threadIdx.x;       // 0..127
  const size_t base = (size_t)t * kQKV + g * (6 * kD) + slot * kD;
  if (slot == 5) {                 // V: no rope, transpose to (G, D, T)
    Vt[((size_t)g * kD + d) * kT + t] = f2bf(qkv[base + d]);
    return;
  }
  const int dh = d & 63;
  const float c = cosb[t * 64 + dh];
  const float s = sinb[t * 64 + dh];
  const float t1 = qkv[base + dh];
  const float t2 = qkv[base + dh + 64];
  const float r = (d < 64) ? (t1 * c - t2 * s) : (t1 * s + t2 * c);
  if (slot == 4) {
    Kq[((size_t)g * kT + t) * kD + d] = f2bf((r - kb[d] > 0.f) ? ka[d] : 0.f);
  } else {
    const int h = g * 4 + slot;
    Q[((size_t)h * kT + t) * kD + d] = f2bf((r - qb[d] > 0.f) ? qa[d] : 0.f);
  }
}

// ------------------------------- WMMA GEMM ---------------------------------
// C(M,N) = A(M,K)bf16 @ B(N,K)bf16^T  (+ optional f32 addsrc), f32 out.
// 128x128 block tile, 8 waves of (64x32), K-step 32, double-buffered LDS
// (2 x 16KB), next tile staged via global_load_async_to_lds_b128 while the
// current tile feeds 8 v_wmma per K-step.
__device__ __forceinline__ void stage_row16(const bf16_t* __restrict__ src,
                                            bf16_t* __restrict__ dst) {
#if HAVE_ASYNC_LDS
  async_copy16(src, dst);
  async_copy16(src + 8, dst + 8);
#else
  ((uint4*)dst)[0] = ((const uint4*)src)[0];
  ((uint4*)dst)[1] = ((const uint4*)src)[1];
#endif
}

__global__ __launch_bounds__(256) void gemm_bf16(
    const bf16_t* __restrict__ A, const bf16_t* __restrict__ Bw,
    float* __restrict__ Cout, const float* __restrict__ addsrc,
    int M, int N, int K) {
  __shared__ __attribute__((aligned(32))) bf16_t As[2][128 * 32];
  __shared__ __attribute__((aligned(32))) bf16_t Bs[2][128 * 32];
  const int tid = threadIdx.x;
  const int lane = tid & 31;
  const int wid = tid >> 5;
  const int wm = wid & 1;          // 2 wave rows x 64
  const int wn = wid >> 1;         // 4 wave cols x 32
  const int m0 = blockIdx.y * 128;
  const int n0 = blockIdx.x * 128;

  const v8f zero = {0.f, 0.f, 0.f, 0.f, 0.f, 0.f, 0.f, 0.f};
  v8f acc[4][2];
#pragma unroll
  for (int i = 0; i < 4; ++i)
#pragma unroll
    for (int j = 0; j < 2; ++j) acc[i][j] = zero;

  const int ldr = tid >> 1;        // 0..127: tile row to stage
  const int ldh = (tid & 1) * 16;  // which 16-element K half
  const int kbA = (lane >> 4) << 4;  // A frag: contiguous 16 K values
  const int kbB = (lane >> 4) << 4;  // B frag: contiguous 16 K values

  const bf16_t* aSrc = A + (size_t)(m0 + ldr) * K + ldh;
  const bf16_t* bSrc = Bw + (size_t)(n0 + ldr) * K + ldh;

  // prologue: stage K-step 0 into buffer 0
  stage_row16(aSrc, &As[0][ldr * 32 + ldh]);
  stage_row16(bSrc, &Bs[0][ldr * 32 + ldh]);

  const int nsteps = K >> 5;
  for (int i = 0; i < nsteps; ++i) {
    const int p = i & 1;
    const bool pre = (i + 1) < nsteps;
    if (pre) {  // prefetch next K slice into the other buffer
      const int k1 = (i + 1) << 5;
      stage_row16(aSrc + k1, &As[1 - p][ldr * 32 + ldh]);
      stage_row16(bSrc + k1, &Bs[1 - p][ldr * 32 + ldh]);
      wait_async_le4();            // current tile's 4 copies done; 4 in flight
    } else {
      wait_async_le0();
    }
    __syncthreads();

    v16bf bfrag[2];
#pragma unroll
    for (int ni = 0; ni < 2; ++ni)
      bfrag[ni] = *(const v16bf*)(&Bs[p][(wn * 32 + ni * 16 + (lane & 15)) * 32 + kbB]);

#pragma unroll
    for (int mi = 0; mi < 4; ++mi) {
      union { v16bf v; v8bf h[2]; } a;
      const bf16_t* pa = &As[p][(wm * 64 + mi * 16 + (lane & 15)) * 32 + kbA];
      a.h[0] = *(const v8bf*)(pa);
      a.h[1] = *(const v8bf*)(pa + 8);
#pragma unroll
      for (int ni = 0; ni < 2; ++ni)
        acc[mi][ni] = __builtin_amdgcn_wmma_f32_16x16x32_bf16(
            false, a.v, false, bfrag[ni], (short)0, acc[mi][ni], false, false);
    }
    __syncthreads();               // all reads of buf p done before it is re-staged
  }

  const int cn = n0 + wn * 32 + (lane & 15);
#pragma unroll
  for (int mi = 0; mi < 4; ++mi) {
    const int rbase = m0 + wm * 64 + mi * 16 + ((lane >> 4) << 3);
#pragma unroll
    for (int ni = 0; ni < 2; ++ni) {
      const int col = cn + ni * 16;
#pragma unroll
      for (int r = 0; r < 8; ++r) {
        const size_t idx = (size_t)(rbase + r) * N + col;
        float v = acc[mi][ni][r];
        if (addsrc) v += addsrc[idx];
        Cout[idx] = v;
      }
    }
  }
}

// --------------------------- causal ReLU attention -------------------------
// One wave per (16-query tile, head). Q/K/Vt read straight from global in
// fragment layout; relu(score)/D relayout D->A via 1KB LDS + s_wait_dscnt.
__global__ __launch_bounds__(32) void attn_relu(
    const bf16_t* __restrict__ Q, const bf16_t* __restrict__ Kb,
    const bf16_t* __restrict__ Vt, float* __restrict__ Y) {
  __shared__ __attribute__((aligned(32))) bf16_t Ps[16 * 32];
  const int lane = threadIdx.x & 31;
  const int t0 = blockIdx.x * 16;
  const int h = blockIdx.y;
  const int g = h >> 2;            // GQA: 4 q heads per kv head
  const int rA = lane & 15;
  const int kbA = (lane >> 4) << 4;
  const int kbB = (lane >> 4) << 4;

  union FragA { v16bf v; v8bf h[2]; };
  FragA qf[4];                     // Q row tile, D=128 -> 4 K-chunks of 32
  const bf16_t* qrow = Q + ((size_t)h * kT + t0 + rA) * kD;
#pragma unroll
  for (int kc = 0; kc < 4; ++kc) {
    qf[kc].h[0] = *(const v8bf*)(qrow + kc * 32 + kbA);
    qf[kc].h[1] = *(const v8bf*)(qrow + kc * 32 + kbA + 8);
  }

  const v8f zero = {0.f, 0.f, 0.f, 0.f, 0.f, 0.f, 0.f, 0.f};
  v8f yacc[8];
#pragma unroll
  for (int i = 0; i < 8; ++i) yacc[i] = zero;

  for (int s0 = 0; s0 <= t0 + 15; s0 += 32) {   // causal: only keys <= query
    v8f sacc[2] = {zero, zero};
#pragma unroll
    for (int nt = 0; nt < 2; ++nt) {
      const bf16_t* krow = Kb + ((size_t)g * kT + s0 + nt * 16 + rA) * kD;
#pragma unroll
      for (int kc = 0; kc < 4; ++kc) {
        v16bf bf = *(const v16bf*)(krow + kc * 32 + kbB);
        sacc[nt] = __builtin_amdgcn_wmma_f32_16x16x32_bf16(
            false, qf[kc].v, false, bf, (short)0, sacc[nt], false, false);
      }
    }
    // mask + relu + 1/D, write P (16 queries x 32 keys) to LDS row-major
#pragma unroll
    for (int nt = 0; nt < 2; ++nt) {
      const int n = nt * 16 + (lane & 15);
#pragma unroll
      for (int r = 0; r < 8; ++r) {
        const int m = ((lane >> 4) << 3) + r;
        float v = sacc[nt][r];
        v = (s0 + n <= t0 + m) ? fmaxf(v, 0.f) * (1.f / 128.f) : 0.f;
        Ps[m * 32 + n] = f2bf(v);
      }
    }
    asm volatile("s_wait_dscnt 0" ::: "memory");  // LDS RAW fence (wave-local)
    FragA pf;
    {
      const bf16_t* p = Ps + rA * 32 + kbA;
      pf.h[0] = *(const v8bf*)(p);
      pf.h[1] = *(const v8bf*)(p + 8);
    }
#pragma unroll
    for (int dt = 0; dt < 8; ++dt) {             // y += P @ V, V^T is (G,D,T)
      const bf16_t* vrow = Vt + ((size_t)g * kD + dt * 16 + rA) * kT + s0 + kbB;
      v16bf vf = *(const v16bf*)vrow;
      yacc[dt] = __builtin_amdgcn_wmma_f32_16x16x32_bf16(
          false, pf.v, false, vf, (short)0, yacc[dt], false, false);
    }
  }
#pragma unroll
  for (int dt = 0; dt < 8; ++dt) {
    const int col = h * kD + dt * 16 + (lane & 15);
    const int rb = t0 + ((lane >> 4) << 3);
#pragma unroll
    for (int r = 0; r < 8; ++r)
      Y[(size_t)(rb + r) * kC + col] = yacc[dt][r];
  }
}

// ------------------------------- launcher ----------------------------------

extern "C" void kernel_launch(void* const* d_in, const int* in_sizes, int n_in,
                              void* d_out, int out_size, void* d_ws, size_t ws_size,
                              hipStream_t stream) {
  const float* x          = (const float*)d_in[0];
  const float* cosb       = (const float*)d_in[1];
  const float* sinb       = (const float*)d_in[2];
  const float* attn_w     = (const float*)d_in[3];
  const float* proj_w     = (const float*)d_in[4];
  const float* proj_scale = (const float*)d_in[5];
  const float* fc_w       = (const float*)d_in[6];
  const float* mlp_proj_w = (const float*)d_in[7];
  const float* mlp_scale  = (const float*)d_in[8];
  const float* in_alpha   = (const float*)d_in[9];
  const float* in_beta    = (const float*)d_in[10];
  const float* q_alpha    = (const float*)d_in[11];
  const float* q_beta     = (const float*)d_in[12];
  const float* k_alpha    = (const float*)d_in[13];
  const float* k_beta     = (const float*)d_in[14];
  const float* out_alpha  = (const float*)d_in[15];
  const float* out_beta   = (const float*)d_in[16];
  const float* act1_alpha = (const float*)d_in[17];
  const float* act1_beta  = (const float*)d_in[18];
  const float* act2_alpha = (const float*)d_in[19];
  const float* act2_beta  = (const float*)d_in[20];
  (void)in_sizes; (void)n_in; (void)out_size; (void)ws_size;

  char* ws = (char*)d_ws;                       // region plan (sequential reuse)
  bf16_t* ACT = (bf16_t*)(ws);                  //  32 MB: bf16 GEMM A operand
  bf16_t* W   = (bf16_t*)(ws + ((size_t)32 << 20));   //  32 MB: bf16 weights
  float*  BUF = (float*) (ws + ((size_t)64 << 20));   //  64 MB: qkv / Y / m
  float*  X1  = (float*) (ws + ((size_t)128 << 20));  //  16 MB: mid residual
  bf16_t* Qb  = (bf16_t*)(ws + ((size_t)144 << 20));  //   8 MB
  bf16_t* Kq  = (bf16_t*)(ws + ((size_t)152 << 20));  //   2 MB
  bf16_t* Vt  = (bf16_t*)(ws + ((size_t)154 << 20));  //   2 MB

  const int nTC = kT * kC;      // 4,194,304
  const int nTI = kT * kI;      // 16,777,216

  // 1) h = hv(x) -> bf16
  hv_to_bf16<<<(nTC + 255) / 256, 256, 0, stream>>>(x, in_alpha, in_beta, ACT, nTC, kC);
  // 2) attn_w -> bf16
  cvt_to_bf16<<<(kQKV * kC + 255) / 256, 256, 0, stream>>>(attn_w, W, kQKV * kC);
  // 3) qkv = h @ attn_w^T : (2048 x 3072)
  gemm_bf16<<<dim3(kQKV / 128, kT / 128), 256, 0, stream>>>(ACT, W, BUF, nullptr, kT, kQKV, kC);
  // 4) rope + gates + split
  rope_split<<<dim3(kT, 24), 128, 0, stream>>>(BUF, cosb, sinb, q_alpha, q_beta,
                                               k_alpha, k_beta, Qb, Kq, Vt);
  // 5) causal-relu attention -> Y (reuses BUF)
  attn_relu<<<dim3(kT / 16, kH), 32, 0, stream>>>(Qb, Kq, Vt, BUF);
  // 6) hv(Y) -> bf16
  hv_to_bf16<<<(nTC + 255) / 256, 256, 0, stream>>>(BUF, out_alpha, out_beta, ACT, nTC, kC);
  // 7) normalize proj_w -> bf16
  norm_weight<<<kC, 256, 0, stream>>>(proj_w, W, kC, kC, proj_scale);
  // 8) X1 = x + hv(Y) @ projW^T
  gemm_bf16<<<dim3(kC / 128, kT / 128), 256, 0, stream>>>(ACT, W, X1, x, kT, kC, kC);
  // 9) hv(X1) -> bf16
  hv_to_bf16<<<(nTC + 255) / 256, 256, 0, stream>>>(X1, act1_alpha, act1_beta, ACT, nTC, kC);
  // 10) fc_w -> bf16
  cvt_to_bf16<<<(kI * kC + 255) / 256, 256, 0, stream>>>(fc_w, W, kI * kC);
  // 11) m = hv(X1) @ fc_w^T : (2048 x 8192)
  gemm_bf16<<<dim3(kI / 128, kT / 128), 256, 0, stream>>>(ACT, W, BUF, nullptr, kT, kI, kC);
  // 12) hv(m) -> bf16
  hv_to_bf16<<<(nTI + 255) / 256, 256, 0, stream>>>(BUF, act2_alpha, act2_beta, ACT, nTI, kI);
  // 13) normalize mlp_proj_w -> bf16
  norm_weight<<<kI, 256, 0, stream>>>(mlp_proj_w, W, kC, kI, mlp_scale);
  // 14) out = X1 + hv(m) @ mlpW^T
  gemm_bf16<<<dim3(kC / 128, kT / 128), 256, 0, stream>>>(ACT, W, (float*)d_out, X1, kT, kC, kI);
}
